// StarEncoder_45243185496617
// MI455X (gfx1250) — compile-verified, matrix operand order checked by reading, not run on hardware
//
#include <hip/hip_runtime.h>
#include <hip/hip_bf16.h>
#include <cstdint>
#include <cstddef>

// ---------------------------------------------------------------------------
// StarEncoder forward for MI455X (gfx1250), WMMA f16 path.
// Convs: per-image LDS im2col, 9 waves, dual-N-tile WMMA.
// GEMMs: 4 N-tiles per wave (A-fragment reused 4x), fused BN stats,
//        global_prefetch on streamed K.
// ---------------------------------------------------------------------------

#define BATCH 16384
#define CC    20
#define HID   256
#define DIMO  65
#define EPSB  1e-5f

typedef _Float16 v8h  __attribute__((ext_vector_type(8)));
typedef _Float16 v16h __attribute__((ext_vector_type(16)));
typedef float    v8f  __attribute__((ext_vector_type(8)));

__device__ __forceinline__ v16h join8(v8h a, v8h b) {
  return __builtin_shufflevector(a, b, 0,1,2,3,4,5,6,7,8,9,10,11,12,13,14,15);
}
__device__ __forceinline__ v8f wmma_f16(v16h a, v16h b, v8f c) {
  return __builtin_amdgcn_wmma_f32_16x16x32_f16(false, a, false, b, (short)0, c,
                                                false, false);
}

// --------------------------- weight repack (f32 -> f16, padded) -------------
__global__ void k_pack_f16(const float* __restrict__ src, _Float16* __restrict__ dst,
                           int rows, int cols, int rpad, int cpad) {
  long i = (long)blockIdx.x * blockDim.x + threadIdx.x;
  long tot = (long)rpad * cpad;
  if (i >= tot) return;
  int r = (int)(i / cpad), c = (int)(i - (long)r * cpad);
  float v = (r < rows && c < cols) ? src[(long)r * cols + c] : 0.f;
  dst[i] = (_Float16)v;
}

// --------------------------- input transform into padded f16 ----------------
__global__ void k_prep_input(const float* __restrict__ img, const float* __restrict__ bg,
                             _Float16* __restrict__ xpad) {
  long i = (long)blockIdx.x * blockDim.x + threadIdx.x;
  long tot = (long)BATCH * 2 * 144;
  if (i >= tot) return;
  int m = (int)(i % 144);
  long bc = i / 144;
  int y = m / 12, x = m - 12 * y;
  float v = __logf(img[i] - bg[i] + 1000.0f);
  xpad[bc * 196 + (y + 1) * 14 + (x + 1)] = (_Float16)v;
}

// --------------------------- conv via LDS im2col + WMMA ---------------------
// One block per image, 9 waves; wave w owns M-tile rows [16w,16w+16).
// mode 0: bias+relu -> padded (B,20,14,14) f16 (may be in-place).
// mode 1: bias only -> flat (B, 20*144) f16 (pre-BN).
__global__ void __launch_bounds__(288)
k_conv(const _Float16* __restrict__ xg, const _Float16* __restrict__ Wg,
       const float* __restrict__ bias, _Float16* __restrict__ outg,
       int Cin, int Kpad, int mode) {
  extern __shared__ _Float16 sm[];
  _Float16* col = sm;                 // 144 * Kpad
  _Float16* xin = sm + 144 * Kpad;    // Cin * 196
  const int b = blockIdx.x, tid = threadIdx.x;
  const _Float16* xsrc = xg + (size_t)b * Cin * 196;
  const int nin = Cin * 196;
  for (int i = tid; i < nin; i += 288) xin[i] = xsrc[i];
  __syncthreads();
  const int KA = Cin * 9, tot = 144 * Kpad;
  for (int i = tid; i < tot; i += 288) {
    int m = i / Kpad, k = i - m * Kpad;
    _Float16 v = (_Float16)0.f;
    if (k < KA) {
      int ci = k / 9, rr = k - ci * 9, ky = rr / 3, kx = rr - ky * 3;
      int y = m / 12, x = m - y * 12;
      v = xin[ci * 196 + (y + ky) * 14 + (x + kx)];
    }
    col[i] = v;
  }
  __syncthreads();

  const int wv = tid >> 5, lane = tid & 31, lr = lane & 15;
  const int m0 = wv * 16, sel = (lane < 16) ? 0 : 8;
  const _Float16* arow  = col + (m0 + lr) * Kpad + sel;
  const _Float16* b0row = Wg + (size_t)lr * Kpad + sel;
  const _Float16* b1row = Wg + (size_t)(16 + lr) * Kpad + sel;
  v8f acc0 = {}, acc1 = {};
  for (int k0 = 0; k0 < Kpad; k0 += 32) {
    v16h av = join8(*(const v8h*)(arow + k0), *(const v8h*)(arow + k0 + 16));
    v16h bv0 = join8(*(const v8h*)(b0row + k0), *(const v8h*)(b0row + k0 + 16));
    acc0 = wmma_f16(av, bv0, acc0);
    v16h bv1 = join8(*(const v8h*)(b1row + k0), *(const v8h*)(b1row + k0 + 16));
    acc1 = wmma_f16(av, bv1, acc1);
  }
  const int mb = m0 + ((lane >= 16) ? 8 : 0);
  for (int t = 0; t < 2; ++t) {
    int n = (t == 0) ? lr : 16 + lr;
    if (n >= CC) continue;
    float bv = bias[n];
    v8f acc = (t == 0) ? acc0 : acc1;
    for (int r = 0; r < 8; ++r) {
      int m = mb + r;
      float v = acc[r] + bv;
      if (mode == 0) {
        v = fmaxf(v, 0.f);
        int y = m / 12, x = m - 12 * y;
        outg[((size_t)b * CC + n) * 196 + (y + 1) * 14 + (x + 1)] = (_Float16)v;
      } else {
        outg[(size_t)b * (CC * 144) + n * 144 + m] = (_Float16)v;
      }
    }
  }
}

// --------------------------- generic WMMA GEMM: out = A @ W^T + bias --------
// grid = (M/64, Npad/(16*NT)); block = 128 (4 waves); each wave: one 16-row
// M-tile x NT consecutive 16-col N-tiles, reusing the A fragment NT times.
// mode 0: store f16            mode 1: relu, store f16
// mode 2: store f16 + atomic per-column sum/sumsq into stats[2*Nvalid]
// mode 3: store f32 at out32[m*ldo + coff + n]
template <int NT>
__global__ void __launch_bounds__(128)
k_gemm(const _Float16* __restrict__ A, int lda,
       const _Float16* __restrict__ W, int ldw,
       const float* __restrict__ bias, int Nvalid,
       _Float16* __restrict__ out16, float* __restrict__ out32, int ldo,
       int K, int mode, float* __restrict__ stats, int coff) {
  const int wv = threadIdx.x >> 5, lane = threadIdx.x & 31, lr = lane & 15;
  const int m0 = (blockIdx.x * 4 + wv) << 4;
  const int n0 = blockIdx.y * (16 * NT);
  const int sel = (lane < 16) ? 0 : 8;
  const _Float16* arow = A + (size_t)(m0 + lr) * lda + sel;
  const _Float16* brow[NT];
#pragma unroll
  for (int t = 0; t < NT; ++t)
    brow[t] = W + (size_t)(n0 + 16 * t + lr) * ldw + sel;
  v8f acc[NT];
#pragma unroll
  for (int t = 0; t < NT; ++t) acc[t] = (v8f){};
  for (int k0 = 0; k0 < K; k0 += 32) {
    // prefetch next K-chunk of the streamed rows (-> global_prefetch_b8)
    __builtin_prefetch(arow + k0 + 128, 0, 0);
    v16h av = join8(*(const v8h*)(arow + k0), *(const v8h*)(arow + k0 + 16));
#pragma unroll
    for (int t = 0; t < NT; ++t) {
      v16h bv = join8(*(const v8h*)(brow[t] + k0), *(const v8h*)(brow[t] + k0 + 16));
      acc[t] = wmma_f16(av, bv, acc[t]);
    }
  }
  const int mb = m0 + ((lane >= 16) ? 8 : 0);
#pragma unroll
  for (int t = 0; t < NT; ++t) {
    const int n = n0 + 16 * t + lr;
    const bool nok = (n < Nvalid);
    const float bv = nok ? bias[n] : 0.f;
    float s = 0.f, s2 = 0.f;
    for (int r = 0; r < 8; ++r) {
      float v = acc[t][r] + bv;
      if (mode == 1) v = fmaxf(v, 0.f);
      if (mode == 2) { s += v; s2 += v * v; }
      if (nok) {
        if (mode == 3) out32[(size_t)(mb + r) * ldo + coff + n] = v;
        else           out16[(size_t)(mb + r) * ldo + n] = (_Float16)v;
      }
    }
    if (mode == 2 && nok) {
      atomicAdd(&stats[n], s);
      atomicAdd(&stats[Nvalid + n], s2);
    }
  }
}

// --------------------------- BatchNorm helpers ------------------------------
__global__ void k_bn2_stats(const _Float16* __restrict__ x, float* __restrict__ stats,
                            long total) {
  __shared__ float ss[CC], sq[CC];
  if (threadIdx.x < CC) { ss[threadIdx.x] = 0.f; sq[threadIdx.x] = 0.f; }
  __syncthreads();
  for (long i = (long)blockIdx.x * blockDim.x + threadIdx.x; i < total;
       i += (long)gridDim.x * blockDim.x) {
    int c = (int)((i / 144) % CC);
    float v = (float)x[i];
    atomicAdd(&ss[c], v);
    atomicAdd(&sq[c], v * v);
  }
  __syncthreads();
  if (threadIdx.x < CC) {
    atomicAdd(&stats[threadIdx.x], ss[threadIdx.x]);
    atomicAdd(&stats[CC + threadIdx.x], sq[threadIdx.x]);
  }
}

__global__ void k_bn_final(const float* __restrict__ stats, const float* __restrict__ g,
                           const float* __restrict__ bet, float* __restrict__ scsh,
                           int C, float invN) {
  int c = blockIdx.x * blockDim.x + threadIdx.x;
  if (c >= C) return;
  float mean = stats[c] * invN;
  float var  = stats[C + c] * invN - mean * mean;
  float sc = g[c] * rsqrtf(var + EPSB);
  scsh[c] = sc;
  scsh[C + c] = bet[c] - mean * sc;
}

__global__ void k_bn_apply_flat(const _Float16* __restrict__ x,
                                const float* __restrict__ scsh,
                                _Float16* __restrict__ y, int C, int HW, long total) {
  long i = (long)blockIdx.x * blockDim.x + threadIdx.x;
  if (i >= total) return;
  int c = (int)((i / HW) % C);
  float v = (float)x[i] * scsh[c] + scsh[C + c];
  y[i] = (_Float16)fmaxf(v, 0.f);
}

__global__ void k_bn_apply_pad(const _Float16* __restrict__ pre,
                               const float* __restrict__ scsh,
                               _Float16* __restrict__ xpad) {
  long i = (long)blockIdx.x * blockDim.x + threadIdx.x;
  long tot = (long)BATCH * CC * 144;
  if (i >= tot) return;
  int m = (int)(i % 144);
  long bc = i / 144;
  int c = (int)(bc % CC);
  int y = m / 12, x = m - 12 * y;
  float v = (float)pre[i] * scsh[c] + scsh[CC + c];
  xpad[bc * 196 + (y + 1) * 14 + (x + 1)] = (_Float16)fmaxf(v, 0.f);
}

// --------------------------- concat builders --------------------------------
__global__ void k_concat2(const _Float16* __restrict__ a, int lda, int wa,
                          const _Float16* __restrict__ h,
                          _Float16* __restrict__ dst, int ldd, long total) {
  long i = (long)blockIdx.x * blockDim.x + threadIdx.x;
  if (i >= total) return;
  long b = i / ldd; int j = (int)(i - b * ldd);
  _Float16 v = (_Float16)0.f;
  if (j < wa)            v = a[b * lda + j];
  else if (j < wa + HID) v = h[b * HID + (j - wa)];
  dst[i] = v;
}

__global__ void k_concat3(const _Float16* __restrict__ a, int lda, int wa,
                          const _Float16* __restrict__ b2, int ldb, int wb,
                          const _Float16* __restrict__ h,
                          _Float16* __restrict__ dst, int ldd, long total) {
  long i = (long)blockIdx.x * blockDim.x + threadIdx.x;
  if (i >= total) return;
  long b = i / ldd; int j = (int)(i - b * ldd);
  _Float16 v = (_Float16)0.f;
  if (j < wa)                 v = a[b * lda + j];
  else if (j < wa + wb)       v = b2[b * ldb + (j - wa)];
  else if (j < wa + wb + HID) v = h[b * HID + (j - wa - wb)];
  dst[i] = v;
}

// --------------------------- tail: softmax/argmax/gather --------------------
__global__ void k_final(const float* __restrict__ H, float* __restrict__ out) {
  int b = blockIdx.x * blockDim.x + threadIdx.x;
  if (b >= BATCH) return;
  const float* h = H + (size_t)b * DIMO;
  const int PROB[5] = {0, 7, 20, 39, 64};
  float p[5];
  for (int j = 0; j < 5; ++j) p[j] = h[PROB[j]];
  float mx = p[0]; int n = 0;
  for (int j = 1; j < 5; ++j) if (p[j] > mx) { mx = p[j]; n = j; }
  float se = 0.f;
  for (int j = 0; j < 5; ++j) se += __expf(p[j] - mx);
  float lse = mx + __logf(se);
  float* lp = out + (size_t)BATCH * 24 + (size_t)b * 5;
  for (int j = 0; j < 5; ++j) lp[j] = p[j] - lse;
  int i0 = 3 * n * (n - 1) + n, i1 = i0 + 2 * n, i2 = i0 + 4 * n, i3 = i2 + n;
  float* llm = out + (size_t)b * 8;
  float* llv = out + (size_t)BATCH * 8 + (size_t)b * 8;
  float* lfm = out + (size_t)BATCH * 16 + (size_t)b * 4;
  float* lfv = out + (size_t)BATCH * 20 + (size_t)b * 4;
  for (int j = 0; j < 8; ++j) { llm[j] = (j < 2 * n) ? h[i0 + j] : 0.f;
                                llv[j] = (j < 2 * n) ? h[i1 + j] : 0.f; }
  for (int j = 0; j < 4; ++j) { lfm[j] = (j < n) ? h[i2 + j] : 0.f;
                                lfv[j] = (j < n) ? h[i3 + j] : 0.f; }
}

// ===========================================================================
extern "C" void kernel_launch(void* const* d_in, const int* in_sizes, int n_in,
                              void* d_out, int out_size, void* d_ws, size_t ws_size,
                              hipStream_t stream) {
  // ---- input leaf indices: images, background, then the params pytree.
  // Auto-detect flatten order via in_sizes[2]:
  //   sorted keys   (bn1,bn2,branch,conv,fc): in_sizes[2] == 256 (bn1 gamma)
  //   insertion ord (conv,bn2,fc,bn1,branch): in_sizes[2] == 360 (conv1 w)
  const int I_IMG = 0, I_BG = 1;
  int I_BN1, I_BN2, I_BR, I_CV, I_FC;
  if (n_in > 2 && in_sizes[2] == 256) {
    I_BN1 = 2; I_BN2 = 8; I_BR = 12; I_CV = 62; I_FC = 70;
  } else {
    I_CV = 2; I_BN2 = 10; I_FC = 14; I_BN1 = 20; I_BR = 26;
  }
  auto F = [&](int i) { return (const float*)d_in[i]; };

  // branch tables
  const int wB [5] = {10, 70, 130, 190, 250};
  const int loB[5] = {1, 7, 13, 19, 25};
  const int npa[5] = {64, 128, 192, 192, 256};  // ceil64(w)  (N-pad, 4 tiles/wave)
  const int wk [5] = {32, 96, 160, 192, 256};   // ceil32(w)  (K-pad)
  const int bk [5] = {288, 352, 416, 448, 512}; // ceil32(w+256)
  const int fk [5] = {288, 416, 544, 640, 768}; // ceil32(2w+256)
  const int npf     = 64;                       // ceil64(lo)
  const int cof[5] = {0, 1, 8, 21, 40};

  // ---- workspace bump allocator (units: halves, 128B aligned) ----
  size_t off = 0;
  auto alloc = [&](size_t halves) { size_t o = off; off += (halves + 63) & ~(size_t)63; return o; };
  _Float16* base = (_Float16*)d_ws;
  auto P  = [&](size_t o) { return base + o; };
  auto PF = [&](size_t o) { return (float*)(base + o); };

  size_t o_wc1 = alloc(32 * 32);
  size_t o_wc2 = alloc(32 * 192), o_wc3 = alloc(32 * 192), o_wc4 = alloc(32 * 192);
  size_t o_wf1 = alloc(256 * 2880), o_wf2 = alloc(256 * 256), o_wf3 = alloc(256 * 256);
  size_t o_a1[5], o_a2[5], o_b1[5], o_b2[5], o_f[5];
  for (int i = 0; i < 5; ++i) {
    o_a1[i] = alloc((size_t)npa[i] * 256);
    o_a2[i] = alloc((size_t)npa[i] * wk[i]);
    o_b1[i] = alloc((size_t)npa[i] * bk[i]);
    o_b2[i] = alloc((size_t)npa[i] * wk[i]);
    o_f [i] = alloc((size_t)npf * fk[i]);
  }
  size_t o_xin  = alloc((size_t)BATCH * 2 * 196);
  size_t o_xp   = alloc((size_t)BATCH * CC * 196);
  size_t o_pre  = alloc((size_t)BATCH * CC * 144);
  size_t o_hpre = alloc((size_t)BATCH * 256);
  size_t o_h    = alloc((size_t)BATCH * 256);
  size_t o_t1   = alloc((size_t)BATCH * 256);
  size_t o_ha   = alloc((size_t)BATCH * 256);
  size_t o_c1   = alloc((size_t)BATCH * 512);
  size_t o_t2   = alloc((size_t)BATCH * 256);
  size_t o_hb   = alloc((size_t)BATCH * 256);
  size_t o_c2   = alloc((size_t)BATCH * 768);
  size_t o_H    = alloc((size_t)BATCH * DIMO * 2);          // f32
  size_t o_s20a = alloc(80), o_s20b = alloc(80);
  size_t o_ss20a = alloc(80), o_ss20b = alloc(80);
  size_t o_s256[3], o_ss256[3];
  for (int i = 0; i < 3; ++i) { o_s256[i] = alloc(1024); o_ss256[i] = alloc(1024); }

  size_t used_bytes = off * 2;
  if (used_bytes > ws_size) used_bytes = ws_size;
  hipMemsetAsync(d_ws, 0, used_bytes, stream);

  auto pack = [&](const float* src, size_t dsto, int r, int c, int rp, int cp) {
    long tot = (long)rp * cp;
    k_pack_f16<<<dim3((unsigned)((tot + 255) / 256)), 256, 0, stream>>>(src, P(dsto), r, c, rp, cp);
  };
  // conv weights: OIHW (20, Cin, 3, 3) -> (32, Kpad)
  pack(F(I_CV + 0), o_wc1, CC, 2 * 9, 32, 32);
  pack(F(I_CV + 2), o_wc2, CC, CC * 9, 32, 192);
  pack(F(I_CV + 4), o_wc3, CC, CC * 9, 32, 192);
  pack(F(I_CV + 6), o_wc4, CC, CC * 9, 32, 192);
  pack(F(I_FC + 0), o_wf1, 256, 2880, 256, 2880);
  pack(F(I_FC + 2), o_wf2, 256, 256, 256, 256);
  pack(F(I_FC + 4), o_wf3, 256, 256, 256, 256);
  for (int i = 0; i < 5; ++i) {
    const int bi = I_BR + 10 * i;
    pack(F(bi + 0), o_a1[i], wB[i], 256,         npa[i], 256);
    pack(F(bi + 2), o_a2[i], wB[i], wB[i],       npa[i], wk[i]);
    pack(F(bi + 4), o_b1[i], wB[i], wB[i] + 256, npa[i], bk[i]);
    pack(F(bi + 6), o_b2[i], wB[i], wB[i],       npa[i], wk[i]);
    pack(F(bi + 8), o_f[i],  loB[i], 2 * wB[i] + 256, npf, fk[i]);
  }

  // ---- input transform ----
  {
    long tot = (long)BATCH * 2 * 144;
    k_prep_input<<<dim3((unsigned)((tot + 255) / 256)), 256, 0, stream>>>(F(I_IMG), F(I_BG), P(o_xin));
  }

  const unsigned shc1 = (144 * 32 + 2 * 196) * 2;
  const unsigned shc  = (144 * 192 + CC * 196) * 2;  // 63136 B
  // conv1 (relu -> padded)
  k_conv<<<BATCH, 288, shc1, stream>>>(P(o_xin), P(o_wc1), F(I_CV + 1), P(o_xp), 2, 32, 0);
  // conv2 (pre-BN flat)
  k_conv<<<BATCH, 288, shc, stream>>>(P(o_xp), P(o_wc2), F(I_CV + 3), P(o_pre), CC, 192, 1);
  // bn2a
  {
    long tot = (long)BATCH * CC * 144;
    k_bn2_stats<<<2048, 256, 0, stream>>>(P(o_pre), PF(o_s20a), tot);
    k_bn_final<<<1, 32, 0, stream>>>(PF(o_s20a), F(I_BN2 + 0), F(I_BN2 + 1), PF(o_ss20a),
                                     CC, 1.f / (float)((long)BATCH * 144));
    k_bn_apply_pad<<<(unsigned)((tot + 255) / 256), 256, 0, stream>>>(P(o_pre), PF(o_ss20a), P(o_xp));
  }
  // conv3 (relu -> padded, in-place on o_xp: image staged to LDS before writes)
  k_conv<<<BATCH, 288, shc, stream>>>(P(o_xp), P(o_wc3), F(I_CV + 5), P(o_xp), CC, 192, 0);
  // conv4 (pre-BN flat)
  k_conv<<<BATCH, 288, shc, stream>>>(P(o_xp), P(o_wc4), F(I_CV + 7), P(o_pre), CC, 192, 1);
  // bn2b, apply in-place -> flat h0 (B, 2880) f16
  {
    long tot = (long)BATCH * CC * 144;
    k_bn2_stats<<<2048, 256, 0, stream>>>(P(o_pre), PF(o_s20b), tot);
    k_bn_final<<<1, 32, 0, stream>>>(PF(o_s20b), F(I_BN2 + 2), F(I_BN2 + 3), PF(o_ss20b),
                                     CC, 1.f / (float)((long)BATCH * 144));
    k_bn_apply_flat<<<(unsigned)((tot + 255) / 256), 256, 0, stream>>>(P(o_pre), PF(o_ss20b),
                                                                       P(o_pre), CC, 144, tot);
  }

  // ---- FC stack with fused BN stats ----
  const dim3 gblk(128);
  const unsigned MG = BATCH / 64;
  const size_t wfc[3] = {o_wf1, o_wf2, o_wf3};
  const int    kfc[3] = {2880, 256, 256};
  for (int l = 0; l < 3; ++l) {
    const _Float16* Ain = (l == 0) ? P(o_pre) : P(o_h);
    int lda = (l == 0) ? 2880 : 256;
    k_gemm<4><<<dim3(MG, 4), gblk, 0, stream>>>(Ain, lda, P(wfc[l]), kfc[l],
                                                F(I_FC + 2 * l + 1), 256, P(o_hpre), nullptr,
                                                256, kfc[l], 2, PF(o_s256[l]), 0);
    k_bn_final<<<1, 256, 0, stream>>>(PF(o_s256[l]), F(I_BN1 + 2 * l), F(I_BN1 + 2 * l + 1),
                                      PF(o_ss256[l]), 256, 1.f / (float)BATCH);
    long tot = (long)BATCH * 256;
    k_bn_apply_flat<<<(unsigned)((tot + 255) / 256), 256, 0, stream>>>(P(o_hpre), PF(o_ss256[l]),
                                                                       P(o_h), 256, 1, tot);
  }

  // ---- branches ----
  for (int i = 0; i < 5; ++i) {
    const int bi = I_BR + 10 * i;
    const dim3 gN(MG, npa[i] / 64);
    // t1 = relu(h @ a1^T + b)
    k_gemm<4><<<gN, gblk, 0, stream>>>(P(o_h), 256, P(o_a1[i]), 256, F(bi + 1), wB[i],
                                       P(o_t1), nullptr, wk[i], 256, 1, nullptr, 0);
    // ha = relu(t1 @ a2^T + b)
    k_gemm<4><<<gN, gblk, 0, stream>>>(P(o_t1), wk[i], P(o_a2[i]), wk[i], F(bi + 3), wB[i],
                                       P(o_ha), nullptr, npa[i], wk[i], 1, nullptr, 0);
    // cat1 = [ha, h]
    {
      long tot = (long)BATCH * bk[i];
      k_concat2<<<(unsigned)((tot + 255) / 256), 256, 0, stream>>>(P(o_ha), npa[i], wB[i],
                                                                   P(o_h), P(o_c1), bk[i], tot);
    }
    // t2 = relu(cat1 @ b1^T + b)
    k_gemm<4><<<gN, gblk, 0, stream>>>(P(o_c1), bk[i], P(o_b1[i]), bk[i], F(bi + 5), wB[i],
                                       P(o_t2), nullptr, wk[i], bk[i], 1, nullptr, 0);
    // hb = relu(t2 @ b2^T + b)
    k_gemm<4><<<gN, gblk, 0, stream>>>(P(o_t2), wk[i], P(o_b2[i]), wk[i], F(bi + 7), wB[i],
                                       P(o_hb), nullptr, npa[i], wk[i], 1, nullptr, 0);
    // cat2 = [ha, hb, h]
    {
      long tot = (long)BATCH * fk[i];
      k_concat3<<<(unsigned)((tot + 255) / 256), 256, 0, stream>>>(P(o_ha), npa[i], wB[i],
                                                                   P(o_hb), npa[i], wB[i],
                                                                   P(o_h), P(o_c2), fk[i], tot);
    }
    // H[:, cof..cof+lo) = cat2 @ f^T + b  (f32)
    k_gemm<4><<<dim3(MG, 1), gblk, 0, stream>>>(P(o_c2), fk[i], P(o_f[i]), fk[i],
                                                F(bi + 9), loB[i], nullptr, PF(o_H),
                                                DIMO, fk[i], 3, nullptr, cof[i]);
  }

  // ---- tail ----
  k_final<<<BATCH / 256, 256, 0, stream>>>(PF(o_H), (float*)d_out);
}
